// React_Reduction_Block_1881195675756
// MI455X (gfx1250) — 8-bit matrix operand order repaired and checked by reading, compiled, not run on hardware
//
#include <hip/hip_runtime.h>
#include <hip/hip_bf16.h>
#include <math.h>

typedef int v8i __attribute__((ext_vector_type(8)));
typedef unsigned int u32;
typedef unsigned long long u64;

#define Bn   32
#define Cn   256
#define Hn   56
#define Wn   56
#define HO   28
#define WO   28
#define M1   (Bn*HO*WO)
#define KC1  36
#define KC2  4
#define ABUF 10240

__device__ __forceinline__ void async_load16(u32 lds, const void* g) {
    asm volatile("global_load_async_to_lds_b128 %0, %1, off"
                 :: "v"(lds), "v"((u64)(uintptr_t)g) : "memory");
}
template<int N>
__device__ __forceinline__ void wait_async() {
    asm volatile("s_wait_asynccnt %0" :: "n"(N) : "memory");
}
__device__ __forceinline__ u32 lds_off(const void* p) {
    return (u32)(uintptr_t)p;
}

__global__ void zero_sums(float* __restrict__ par) {
    int t = threadIdx.x;
    for (int i = t; i < 1024; i += 256) par[512 + i] = 0.f;
    if (t < 16) par[2560 + t] = 0.f;
}

__global__ __launch_bounds__(256) void binarize1(
    const float* __restrict__ x, const float* __restrict__ beta,
    signed char* __restrict__ xs)
{
    __shared__ float T[56][65];
    int cg = blockIdx.x, h = blockIdx.y, b = blockIdx.z;
    int tid = threadIdx.x;
    int r = tid >> 2, seg = tid & 3;
    const float* src = x + (((size_t)(b*Cn + cg*64 + r)*Hn + h)*Wn);
    #pragma unroll
    for (int i = 0; i < 14; ++i) {
        int w = seg*14 + i;
        T[w][r] = src[w];
    }
    __syncthreads();
    int c = tid & 63, wg = tid >> 6;
    float bb = beta[cg*64 + c];
    signed char* dst = xs + ((size_t)(b*Hn + h)*Wn)*Cn + cg*64 + c;
    #pragma unroll
    for (int i = 0; i < 14; ++i) {
        int w = wg*14 + i;
        float v = T[w][c] + bb;
        dst[(size_t)w*Cn] = (v > 0.f) ? 1 : ((v < 0.f) ? -1 : 0);
    }
}

__global__ __launch_bounds__(256) void prep_alpha(
    const float* __restrict__ w1, const float* __restrict__ w2,
    float* __restrict__ alpha1, float* __restrict__ alpha2)
{
    __shared__ float red[256];
    int blk = blockIdx.x, tid = threadIdx.x;
    float s = 0.f;
    if (blk < 256) {
        const float* w = w1 + (size_t)blk * (Cn*9);
        for (int i = tid; i < Cn*9; i += 256) s += fabsf(w[i]);
    } else {
        const float* w = w2 + (size_t)(blk-256) * Cn;
        s = fabsf(w[tid]);
    }
    red[tid] = s; __syncthreads();
    for (int off = 128; off > 0; off >>= 1) {
        if (tid < off) red[tid] += red[tid+off];
        __syncthreads();
    }
    if (tid == 0) {
        if (blk < 256) alpha1[blk] = red[0] / (float)(Cn*9);
        else           alpha2[blk-256] = red[0] / (float)Cn;
    }
}

__global__ void pack_w1(const float* __restrict__ w1, u32* __restrict__ bp) {
    int idx = blockIdx.x*256 + threadIdx.x;
    int v = idx & 7, rest = idx >> 3;
    int lane = rest & 31; rest >>= 5;
    int nt = rest & 15; int kc = rest >> 4;
    int n = nt*16 + (lane & 15);
    int kbase = kc*64 + (v>>2)*32 + (lane>>4)*16 + (v&3)*4;
    u32 d = 0;
    #pragma unroll
    for (int tt = 0; tt < 4; ++tt) {
        int k = kbase + tt;
        int t = k >> 8, ci = k & 255;
        float w = w1[((size_t)n*Cn + ci)*9 + t];
        signed char sg = (w > 0.f) ? 1 : ((w < 0.f) ? -1 : 0);
        d |= ((u32)(unsigned char)sg) << (8*tt);
    }
    bp[idx] = d;
}

__global__ void pack_w2(const float* __restrict__ w2, u32* __restrict__ bp) {
    int idx = blockIdx.x*256 + threadIdx.x;
    int v = idx & 7, rest = idx >> 3;
    int lane = rest & 31; rest >>= 5;
    int nt = rest & 15; int kc = rest >> 4;
    int n = nt*16 + (lane & 15);
    int kbase = kc*64 + (v>>2)*32 + (lane>>4)*16 + (v&3)*4;
    u32 d = 0;
    #pragma unroll
    for (int tt = 0; tt < 4; ++tt) {
        int ci = kbase + tt;
        float w = w2[(size_t)n*Cn + ci];
        signed char sg = (w > 0.f) ? 1 : ((w < 0.f) ? -1 : 0);
        d |= ((u32)(unsigned char)sg) << (8*tt);
    }
    bp[idx] = d;
}

template<int MODE>
__global__ __launch_bounds__(256) void conv_gemm(
    const signed char* __restrict__ asrc, const u32* __restrict__ bpack,
    const float* __restrict__ alpha, const signed char* __restrict__ zpage,
    float* __restrict__ yraw, float* __restrict__ gsum, float* __restrict__ gsumsq)
{
    __shared__ __align__(16) unsigned char Ald[4*ABUF];
    __shared__ float lsum[128], lsq[128];

    const int KC = (MODE == 0) ? KC1 : KC2;
    int tid = threadIdx.x;
    int lane = tid & 31, wid = tid >> 5;
    int wm = wid & 3, wn = wid >> 2;
    int mtile = blockIdx.x, ntile = blockIdx.y;

    int soh[2], sow[2];
    u32 ldso[2];
    const signed char* sb[2];
    #pragma unroll
    for (int it = 0; it < 2; ++it) {
        int linear = tid + it*256;
        int pix = linear >> 2, seg = linear & 3;
        int m = mtile*128 + pix;
        int b = m / (HO*WO), r = m - b*(HO*WO);
        soh[it] = r / WO; sow[it] = r - soh[it]*WO;
        ldso[it] = lds_off(Ald + pix*80 + seg*16);
        if (MODE == 0) sb[it] = asrc + (size_t)b*Hn*Wn*Cn + seg*16;
        else           sb[it] = asrc + (size_t)m*Cn + seg*16;
    }

    auto issue = [&](int kc, int bs) {
        #pragma unroll
        for (int it = 0; it < 2; ++it) {
            const signed char* g;
            if (MODE == 0) {
                int t = kc >> 2, c = kc & 3;
                int kh = (t*11) >> 5, kw = t - kh*3;
                int ih = soh[it]*2 - 1 + kh, iw = sow[it]*2 - 1 + kw;
                bool ok = ((unsigned)ih < (unsigned)Hn) & ((unsigned)iw < (unsigned)Wn);
                g = ok ? (sb[it] + ((ih*Wn + iw)*Cn + c*64)) : (const signed char*)zpage;
            } else {
                g = sb[it] + kc*64;
            }
            async_load16(ldso[it] + bs*ABUF, g);
        }
    };

    v8i zero = {0,0,0,0,0,0,0,0};
    v8i acc[2][4];
    #pragma unroll
    for (int i = 0; i < 2; ++i)
        #pragma unroll
        for (int j = 0; j < 4; ++j) acc[i][j] = zero;

    auto compute = [&](int kc, int bs) {
        v8i bf[4];
        #pragma unroll
        for (int j = 0; j < 4; ++j) {
            int g16 = ntile*8 + wn*4 + j;
            bf[j] = *(const v8i*)(bpack + ((size_t)(kc*16 + g16)*32 + lane)*8);
        }
        #pragma unroll
        for (int i = 0; i < 2; ++i) {
            int row = wm*32 + i*16 + (lane & 15);
            const unsigned char* base = Ald + bs*ABUF + row*80;
            int hi8 = (lane >> 4) * 16;
            v8i af;
            #pragma unroll
            for (int v = 0; v < 8; ++v) {
                int col = (v>>2)*32 + hi8 + (v&3)*4;
                af[v] = *(const u32*)(base + col);
            }
            #pragma unroll
            for (int j = 0; j < 4; ++j)
                acc[i][j] = __builtin_amdgcn_wmma_i32_16x16x64_iu8(
                    true, af, true, bf[j], acc[i][j], false, false);
        }
    };

    issue(0, 0); issue(1, 1);
    for (int kc = 0; kc < KC-2; ++kc) {
        issue(kc+2, (kc+2)&3);
        wait_async<4>();
        __syncthreads();
        compute(kc, kc&3);
    }
    wait_async<2>(); __syncthreads(); compute(KC-2, (KC-2)&3);
    wait_async<0>(); __syncthreads(); compute(KC-1, (KC-1)&3);

    float psum[4] = {0,0,0,0}, psq[4] = {0,0,0,0};
    #pragma unroll
    for (int i = 0; i < 2; ++i) {
        #pragma unroll
        for (int j = 0; j < 4; ++j) {
            int n = ntile*128 + wn*64 + j*16 + (lane & 15);
            float a = alpha[n];
            #pragma unroll
            for (int g = 0; g < 8; ++g) {
                int m = mtile*128 + wm*32 + i*16 + (lane>>4)*8 + g;
                float val = (float)acc[i][j][g] * a;
                yraw[(size_t)m*Cn + n] = val;
                psum[j] += val; psq[j] += val*val;
            }
        }
    }
    if (tid < 128) { lsum[tid] = 0.f; lsq[tid] = 0.f; }
    __syncthreads();
    #pragma unroll
    for (int j = 0; j < 4; ++j) {
        int nloc = wn*64 + j*16 + (lane & 15);
        atomicAdd(&lsum[nloc], psum[j]);
        atomicAdd(&lsq[nloc],  psq[j]);
    }
    __syncthreads();
    if (tid < 128) {
        atomicAdd(&gsum[ntile*128 + tid],   lsum[tid]);
        atomicAdd(&gsumsq[ntile*128 + tid], lsq[tid]);
    }
}

__global__ void bn_finalize(const float* __restrict__ sum, const float* __restrict__ sumsq,
                            const float* __restrict__ gamma, const float* __restrict__ beta,
                            float* __restrict__ scale, float* __restrict__ bias)
{
    int c = threadIdx.x;
    const float inv = 1.f / (float)M1;
    float m = sum[c] * inv;
    float v = sumsq[c] * inv - m*m;
    float sc = gamma[c] * rsqrtf(v + 1e-5f);
    scale[c] = sc;
    bias[c]  = beta[c] - m*sc;
}

__global__ __launch_bounds__(256) void epilogue1(
    const float* __restrict__ x, const float* __restrict__ y1,
    const float* __restrict__ scale1, const float* __restrict__ bias1,
    const float* __restrict__ prg, const float* __restrict__ prz,
    const float* __restrict__ prs, const float* __restrict__ beta2,
    float* __restrict__ out1, signed char* __restrict__ xs2)
{
    __shared__ float P[256][29];
    int blk = blockIdx.x;
    int b = blk / HO, oh = blk - b*HO;
    int tid = threadIdx.x, lane = tid & 31, w = tid >> 5;
    if (lane < 28) {
        #pragma unroll 4
        for (int k = 0; k < 32; ++k) {
            int c = w + 8*k;
            const float* r0 = x + (((size_t)(b*Cn + c)*Hn + 2*oh)*Wn + 2*lane);
            P[c][lane] = 0.25f * (r0[0] + r0[1] + r0[Wn] + r0[Wn+1]);
        }
    }
    __syncthreads();
    int c = tid;
    float sc = scale1[c], bi = bias1[c];
    float g = prg[c], z = prz[c], sl = prs[c], b2 = beta2[c];
    size_t mbase = ((size_t)b*HO + oh)*WO;
    for (int ow = 0; ow < WO; ++ow) {
        size_t m = mbase + ow;
        float val = y1[m*Cn + c]*sc + bi + P[c][ow];
        float xsv = val - g;
        float o = ((xsv > 0.f) ? xsv : sl*xsv) + z;
        out1[m*Cn + c] = o;
        float sg = o + b2;
        xs2[m*Cn + c] = (sg > 0.f) ? 1 : ((sg < 0.f) ? -1 : 0);
    }
}

__global__ __launch_bounds__(256) void final_kernel(
    const float* __restrict__ y2, const float* __restrict__ out1,
    const float* __restrict__ scale2, const float* __restrict__ bias2,
    float* __restrict__ out)
{
    __shared__ float T[256][29];
    int blk = blockIdx.x;
    int b = blk / HO, oh = blk - b*HO;
    int tid = threadIdx.x, lane = tid & 31, w = tid >> 5;
    int c = tid;
    float sc = scale2[c], bi = bias2[c];
    size_t mbase = ((size_t)b*HO + oh)*WO;
    for (int ow = 0; ow < WO; ++ow) {
        size_t m = mbase + ow;
        T[c][ow] = y2[m*Cn + c]*sc + bi + out1[m*Cn + c];
    }
    __syncthreads();
    if (lane < 28) {
        #pragma unroll 4
        for (int k = 0; k < 32; ++k) {
            int co = w + 8*k;
            float v = T[co][lane];
            size_t o0 = (((size_t)b*(2*Cn) + co)*HO + oh)*WO + lane;
            out[o0] = v;
            out[o0 + (size_t)Cn*HO*WO] = v;
        }
    }
}

extern "C" void kernel_launch(void* const* d_in, const int* in_sizes, int n_in,
                              void* d_out, int out_size, void* d_ws, size_t ws_size,
                              hipStream_t stream)
{
    const float* x   = (const float*)d_in[0];
    const float* rb1 = (const float*)d_in[1];
    const float* w1  = (const float*)d_in[2];
    const float* g1  = (const float*)d_in[3];
    const float* be1 = (const float*)d_in[4];
    const float* rb2 = (const float*)d_in[5];
    const float* w2  = (const float*)d_in[6];
    const float* g2  = (const float*)d_in[7];
    const float* be2 = (const float*)d_in[8];
    const float* prg = (const float*)d_in[9];
    const float* prz = (const float*)d_in[10];
    const float* prs = (const float*)d_in[11];
    float* out = (float*)d_out;
    char* ws = (char*)d_ws;

    const size_t OFF_XSIGN = 0;
    const size_t OFF_BP1   = 25690112;
    const size_t OFF_BP2   = 26279936;
    const size_t OFF_PAR   = 26345472;
    const size_t OFF_Y1    = 26361856;
    const size_t OFF_OUT1  = 52051968;
    const size_t OFF_XS2   = 77742080;

    signed char* xsign1 = (signed char*)(ws + OFF_XSIGN);
    u32* bp1 = (u32*)(ws + OFF_BP1);
    u32* bp2 = (u32*)(ws + OFF_BP2);
    float* par = (float*)(ws + OFF_PAR);
    float* alpha1 = par;           float* alpha2 = par + 256;
    float* sum1 = par + 512;       float* sq1 = par + 768;
    float* sum2 = par + 1024;      float* sq2 = par + 1280;
    float* sc1  = par + 1536;      float* bi1 = par + 1792;
    float* sc2  = par + 2048;      float* bi2 = par + 2304;
    signed char* zpage = (signed char*)(par + 2560);
    float* y1raw = (float*)(ws + OFF_Y1);
    float* out1  = (float*)(ws + OFF_OUT1);
    signed char* xsign2 = (signed char*)(ws + OFF_XS2);
    float* y2raw = (float*)(ws + OFF_XSIGN);

    zero_sums<<<1, 256, 0, stream>>>(par);
    binarize1<<<dim3(4, Hn, Bn), 256, 0, stream>>>(x, rb1, xsign1);
    prep_alpha<<<512, 256, 0, stream>>>(w1, w2, alpha1, alpha2);
    pack_w1<<<576, 256, 0, stream>>>(w1, bp1);
    pack_w2<<<64, 256, 0, stream>>>(w2, bp2);
    conv_gemm<0><<<dim3(M1/128, 2), 256, 0, stream>>>(xsign1, bp1, alpha1, zpage, y1raw, sum1, sq1);
    bn_finalize<<<1, 256, 0, stream>>>(sum1, sq1, g1, be1, sc1, bi1);
    epilogue1<<<Bn*HO, 256, 0, stream>>>(x, y1raw, sc1, bi1, prg, prz, prs, rb2, out1, xsign2);
    conv_gemm<1><<<dim3(M1/128, 2), 256, 0, stream>>>(xsign2, bp2, alpha2, zpage, y2raw, sum2, sq2);
    bn_finalize<<<1, 256, 0, stream>>>(sum2, sq2, g2, be2, sc2, bi2);
    final_kernel<<<Bn*HO, 256, 0, stream>>>(y2raw, out1, sc2, bi2, out);
}
